// CliffordRollingAttention_63050119905979
// MI455X (gfx1250) — matrix operand order repaired and checked by reading, not compiled, not run on hardware
//
#include <hip/hip_runtime.h>
#include <hip/hip_bf16.h>

// ---------------------------------------------------------------------------
// CliffordRollingAttention on MI455X (gfx1250, wave32, WMMA + async-to-LDS)
//
// Pipeline:
//   0)    one-shot f32->bf16 conversion of x and w{q,k,v,o}
//   1..3) q_raw/k_raw/v = x @ w{q,k,v}.T + b   (WMMA bf16 GEMM, f32 accum,
//          double-buffered GLOBAL_LOAD_ASYNC_TO_LDS_B128 tile fills)
//   4)    qmix = channel-roll-mix(rmsnorm(q_raw))   [folds the 5 score mixes]
//   5)    k    = rmsnorm(k_raw)                      (in place)
//   6)    attn over 16 sequence shifts -> bf16 activation buffer
//   7)    out  = attn @ wo.T + bo               (WMMA bf16 GEMM -> f32 d_out)
// ---------------------------------------------------------------------------

#define BATCH   2
#define SEQLEN  4096
#define DMODEL  2048
#define NHEADS  16
#define DHEAD   128
#define NROWS   (BATCH * SEQLEN)       // 8192
#define NSHIFT  16

typedef __attribute__((ext_vector_type(16))) __bf16 v16bf;
typedef __attribute__((ext_vector_type(8)))  float  v8f;

__device__ __constant__ int c_seq_shifts[NSHIFT] = {
    0, 1, -1, 3, -3, 9, -9, 26, -26, 78, -78, 232, -232, 689, -689, 2048};

__device__ __forceinline__ unsigned int f2bf(float f) {
    unsigned int u = __float_as_uint(f);
    // round-to-nearest-even truncation to bf16
    u = (u + 0x7FFFu + ((u >> 16) & 1u)) >> 16;
    return u & 0xFFFFu;
}

union FragBF {
    v16bf v;
    uint4 u[2];
};

// Async DMA: copy 16B global -> LDS, ASYNCcnt-tracked (ISA §15.18.3 op 98).
// INST_OFFSET is added to BOTH the global and the LDS address (ISA §10.7),
// so one (lds_off, g_off) pair serves consecutive 16B chunks.
#define ASYNC_B128(lds_off, g_off, s_base, imm)                               \
    asm volatile("global_load_async_to_lds_b128 %0, %1, %2 offset:" #imm      \
                 :: "v"(lds_off), "v"(g_off), "s"(s_base) : "memory")

#define WAIT_ASYNC0() asm volatile("s_wait_asynccnt 0" ::: "memory")

// ---------------------------------------------------------------------------
// One-shot f32 -> bf16 conversion (8 elements / thread, fully coalesced).
// ---------------------------------------------------------------------------
__global__ __launch_bounds__(256) void f32_to_bf16_kernel(
    const float* __restrict__ src, unsigned short* __restrict__ dst) {
    const size_t i = ((size_t)blockIdx.x * 256 + threadIdx.x) * 8;
    const float4 a = *(const float4*)(src + i);
    const float4 b = *(const float4*)(src + i + 4);
    uint4 o;
    o.x = f2bf(a.x) | (f2bf(a.y) << 16);
    o.y = f2bf(a.z) | (f2bf(a.w) << 16);
    o.z = f2bf(b.x) | (f2bf(b.y) << 16);
    o.w = f2bf(b.z) | (f2bf(b.w) << 16);
    *(uint4*)(dst + i) = o;
}

// ---------------------------------------------------------------------------
// Tiled WMMA GEMM:  C[M,N] = A[M,K] * W[N,K]^T + bias[N]   (bf16 in, f32 out)
// Block tile 128x128, K-step 32, 8 waves each owning a 32x64 subtile
// (2x4 grid of 16x16 v_wmma_f32_16x16x32_bf16 accumulators).
// LDS tiles: stride 40 ushorts (80 B) -> 16B-aligned rows AND conflict-free
// 16-row fragment reads. Double-buffered via async-to-LDS DMA so HBM/L2
// fetch of tile k+1 overlaps the WMMA work on tile k.
// ---------------------------------------------------------------------------
__global__ __launch_bounds__(256) void gemm_wmma_bf16(
    const unsigned short* __restrict__ A, const unsigned short* __restrict__ W,
    const float* __restrict__ bias, float* __restrict__ C,
    int M, int N, int K) {
    __shared__ unsigned short lds_a[2][128 * 40];
    __shared__ unsigned short lds_b[2][128 * 40];

    const int t    = threadIdx.x;
    const int lane = t & 31;
    const int wave = t >> 5;
    const int wm   = wave >> 1;      // 0..3  (M direction)
    const int wn   = wave & 1;       // 0..1  (N direction)
    const int half = lane >> 4;      // lane group (K-half select)
    const int ln15 = lane & 15;

    const int m0 = blockIdx.y * 128;
    const int n0 = blockIdx.x * 128;

    // Tile fill assignment: 2 threads per row, each moves 2x16B (32 bf16/row)
    const int frow = t >> 1;
    const int fcol = (t & 1) * 16;   // element offset within 32-wide K tile

    const unsigned lda_off =
        (unsigned)(size_t)(&lds_a[0][0]) + (unsigned)(frow * 40 + fcol) * 2u;
    const unsigned ldb_off =
        (unsigned)(size_t)(&lds_b[0][0]) + (unsigned)(frow * 40 + fcol) * 2u;
    const unsigned lds_buf_bytes = 128u * 40u * 2u;

    const unsigned ga_row = (unsigned)(((size_t)(m0 + frow) * K + fcol) * 2u);
    const unsigned gb_row = (unsigned)(((size_t)(n0 + frow) * K + fcol) * 2u);

    auto issue_tile = [&](int kt, int buf) {
        const unsigned kbytes = (unsigned)(kt * 32) * 2u;
        const unsigned la = lda_off + (unsigned)buf * lds_buf_bytes;
        const unsigned lb = ldb_off + (unsigned)buf * lds_buf_bytes;
        const unsigned ga = ga_row + kbytes;
        const unsigned gb = gb_row + kbytes;
        ASYNC_B128(la, ga, A, 0);
        ASYNC_B128(la, ga, A, 16);
        ASYNC_B128(lb, gb, W, 0);
        ASYNC_B128(lb, gb, W, 16);
    };

    v8f acc[2][4];
#pragma unroll
    for (int i = 0; i < 2; ++i)
#pragma unroll
        for (int j = 0; j < 4; ++j)
#pragma unroll
            for (int r = 0; r < 8; ++r) acc[i][j][r] = 0.0f;

    const int NT = K / 32;
    issue_tile(0, 0);
    WAIT_ASYNC0();
    __syncthreads();

    for (int kt = 0; kt < NT; ++kt) {
        const int cur = kt & 1;
        if (kt + 1 < NT) issue_tile(kt + 1, cur ^ 1);

        FragBF fa[2], fb[4];
#pragma unroll
        for (int i = 0; i < 2; ++i) {
            // A 16x32 bf16 frag: lanes 0-15 -> K 0..7 & 16..23,
            //                    lanes 16-31 -> K 8..15 & 24..31 (ISA 7.12.2)
            const unsigned short* pa =
                &lds_a[cur][(wm * 32 + i * 16 + ln15) * 40 + half * 8];
            fa[i].u[0] = *(const uint4*)pa;
            fa[i].u[1] = *(const uint4*)(pa + 16);
        }
#pragma unroll
        for (int j = 0; j < 4; ++j) {
            // B 32x16 bf16 frag: lanes 0-15 -> K 0..7 & 16..23,
            //                    lanes 16-31 -> K 8..15 & 24..31 (same as A).
            const unsigned short* pb =
                &lds_b[cur][(wn * 64 + j * 16 + ln15) * 40 + half * 8];
            fb[j].u[0] = *(const uint4*)pb;
            fb[j].u[1] = *(const uint4*)(pb + 16);
        }
#pragma unroll
        for (int i = 0; i < 2; ++i)
#pragma unroll
            for (int j = 0; j < 4; ++j)
                acc[i][j] = __builtin_amdgcn_wmma_f32_16x16x32_bf16(
                    false, fa[i].v, false, fb[j].v, (short)0, acc[i][j],
                    false, false);

        if (kt + 1 < NT) {
            WAIT_ASYNC0();
            __syncthreads();
        }
    }

    // Epilogue: C/D layout -> lanes 0-15: M=r, lanes 16-31: M=8+r, N=ln15
#pragma unroll
    for (int i = 0; i < 2; ++i) {
        const int mrow = m0 + wm * 32 + i * 16 + half * 8;
#pragma unroll
        for (int j = 0; j < 4; ++j) {
            const int col = n0 + wn * 64 + j * 16 + ln15;
            const float bv = bias[col];
#pragma unroll
            for (int r = 0; r < 8; ++r)
                C[(size_t)(mrow + r) * N + col] = acc[i][j][r] + bv;
        }
    }
}

// ---------------------------------------------------------------------------
// Row-wise RMSNorm (one block per row of 2048).
// QMIX=true additionally folds the 5 channel-roll score mixes into the row:
//   qmix[d] = sum_n mix[n] * qn[head*128 + ((d - c_n) & 127)],  qn = x*rn*w
// ---------------------------------------------------------------------------
template <bool QMIX>
__global__ __launch_bounds__(256) void rmsnorm_kernel(
    float* __restrict__ data, const float* __restrict__ w,
    const float* __restrict__ mixp) {
    __shared__ float row[DMODEL];
    __shared__ float red[8];

    const size_t base = (size_t)blockIdx.x * DMODEL;
    const int t = threadIdx.x;

    float ss = 0.0f;
#pragma unroll
    for (int i = 0; i < 8; ++i) {
        float x = data[base + t + i * 256];
        row[t + i * 256] = x;
        ss += x * x;
    }
#pragma unroll
    for (int off = 16; off > 0; off >>= 1) ss += __shfl_xor(ss, off, 32);
    if ((t & 31) == 0) red[t >> 5] = ss;
    __syncthreads();

    float tot = 0.0f;
#pragma unroll
    for (int i = 0; i < 8; ++i) tot += red[i];
    const float rn = rsqrtf(tot * (1.0f / (float)DMODEL) + 1e-6f);

    float m0 = 0, m1 = 0, m2 = 0, m3 = 0, m4 = 0;
    if (QMIX) {
        m0 = mixp[0]; m1 = mixp[1]; m2 = mixp[2]; m3 = mixp[3]; m4 = mixp[4];
    }

#pragma unroll
    for (int i = 0; i < 8; ++i) {
        const int d = t + i * 256;
        float y;
        if (QMIX) {
            const int hb = d & ~127;
            const int dd = d & 127;
            const int j1 = hb + ((dd - 1) & 127);
            const int j2 = hb + ((dd - 2) & 127);
            const int j3 = hb + ((dd - 4) & 127);
            const int j4 = hb + ((dd - 8) & 127);
            y = m0 * row[d]  * w[d]  + m1 * row[j1] * w[j1] +
                m2 * row[j2] * w[j2] + m3 * row[j3] * w[j3] +
                m4 * row[j4] * w[j4];
            y *= rn;
        } else {
            y = row[d] * rn * w[d];
        }
        data[base + d] = y;
    }
}

// ---------------------------------------------------------------------------
// Rolling attention: one wave32 per (b, l, h).
//   scores[s] = scale * <qmix[b,l,h,:], k[b,(l-s)&4095,h,:]>
//   softmax over 16 shifts, out = sum_s attn[s] * v[b,(l-s)&4095,h,:]
// Output written as bf16 for the final WMMA GEMM. k/v are L2-resident (64MB).
// ---------------------------------------------------------------------------
__global__ __launch_bounds__(256) void attn_kernel(
    const float* __restrict__ qmix, const float* __restrict__ kbuf,
    const float* __restrict__ vbuf, unsigned short* __restrict__ outbf) {
    const int gid  = blockIdx.x * 8 + (threadIdx.x >> 5);
    const int lane = threadIdx.x & 31;
    const int h = gid & (NHEADS - 1);
    const int l = (gid >> 4) & (SEQLEN - 1);
    const int b = gid >> 16;

    const float scale = 0.08838834764831845f;  // 1/sqrt(128)
    const size_t rowbase =
        ((size_t)(b * SEQLEN + l) * DMODEL) + h * DHEAD + lane * 4;

    const float4 qv = *(const float4*)(qmix + rowbase);

    float sc[NSHIFT];
#pragma unroll
    for (int s = 0; s < NSHIFT; ++s) {
        const int ls = (l - c_seq_shifts[s]) & (SEQLEN - 1);
        const size_t kb =
            ((size_t)(b * SEQLEN + ls) * DMODEL) + h * DHEAD + lane * 4;
        const float4 kv = *(const float4*)(kbuf + kb);
        float d = qv.x * kv.x + qv.y * kv.y + qv.z * kv.z + qv.w * kv.w;
#pragma unroll
        for (int off = 16; off > 0; off >>= 1) d += __shfl_xor(d, off, 32);
        sc[s] = d * scale;
    }

    float mx = sc[0];
#pragma unroll
    for (int s = 1; s < NSHIFT; ++s) mx = fmaxf(mx, sc[s]);
    float wgt[NSHIFT];
    float denom = 0.0f;
#pragma unroll
    for (int s = 0; s < NSHIFT; ++s) {
        wgt[s] = __expf(sc[s] - mx);
        denom += wgt[s];
    }
    const float inv = 1.0f / denom;

    float4 o = {0.0f, 0.0f, 0.0f, 0.0f};
#pragma unroll
    for (int s = 0; s < NSHIFT; ++s) {
        const int ls = (l - c_seq_shifts[s]) & (SEQLEN - 1);
        const size_t vb =
            ((size_t)(b * SEQLEN + ls) * DMODEL) + h * DHEAD + lane * 4;
        const float4 vv = *(const float4*)(vbuf + vb);
        const float a = wgt[s] * inv;
        o.x += a * vv.x; o.y += a * vv.y; o.z += a * vv.z; o.w += a * vv.w;
    }

    uint2 p;
    p.x = f2bf(o.x) | (f2bf(o.y) << 16);
    p.y = f2bf(o.z) | (f2bf(o.w) << 16);
    *(uint2*)(outbf + rowbase) = p;
}

// ---------------------------------------------------------------------------
extern "C" void kernel_launch(void* const* d_in, const int* in_sizes, int n_in,
                              void* d_out, int out_size, void* d_ws,
                              size_t ws_size, hipStream_t stream) {
    (void)in_sizes; (void)n_in; (void)out_size; (void)ws_size;

    const float* x   = (const float*)d_in[0];
    const float* wq  = (const float*)d_in[1];
    const float* bq  = (const float*)d_in[2];
    const float* wk  = (const float*)d_in[3];
    const float* bk  = (const float*)d_in[4];
    const float* wv  = (const float*)d_in[5];
    const float* bv  = (const float*)d_in[6];
    const float* qnw = (const float*)d_in[7];
    const float* knw = (const float*)d_in[8];
    const float* mix = (const float*)d_in[9];
    const float* wo  = (const float*)d_in[10];
    const float* bo  = (const float*)d_in[11];
    float* out = (float*)d_out;

    // Workspace layout (256 MB total):
    //   [ 0, 32M)  x_bf16            -- dead after QKV GEMMs, re-used as the
    //                                   bf16 attention-output buffer
    //   [32M, 64M) wq/wk/wv/wo bf16  (8M each)
    //   [64M,128M) q_raw/qmix  f32
    //   [128M,192M) k_raw/k    f32
    //   [192M,256M) v          f32
    char* ws = (char*)d_ws;
    const size_t XBF  = (size_t)NROWS * DMODEL * sizeof(unsigned short);
    const size_t WBF  = (size_t)DMODEL * DMODEL * sizeof(unsigned short);
    const size_t FBUF = (size_t)NROWS * DMODEL * sizeof(float);

    unsigned short* xbf    = (unsigned short*)ws;
    unsigned short* attnbf = (unsigned short*)ws;           // alias (see above)
    unsigned short* wqbf   = (unsigned short*)(ws + XBF);
    unsigned short* wkbf   = (unsigned short*)(ws + XBF + WBF);
    unsigned short* wvbf   = (unsigned short*)(ws + XBF + 2 * WBF);
    unsigned short* wobf   = (unsigned short*)(ws + XBF + 3 * WBF);
    float* qbuf = (float*)(ws + XBF + 4 * WBF);
    float* kbuf = (float*)(ws + XBF + 4 * WBF + FBUF);
    float* vbuf = (float*)(ws + XBF + 4 * WBF + 2 * FBUF);

    // 0) one-shot bf16 conversions
    f32_to_bf16_kernel<<<(NROWS * DMODEL) / 2048, 256, 0, stream>>>(x, xbf);
    f32_to_bf16_kernel<<<(DMODEL * DMODEL) / 2048, 256, 0, stream>>>(wq, wqbf);
    f32_to_bf16_kernel<<<(DMODEL * DMODEL) / 2048, 256, 0, stream>>>(wk, wkbf);
    f32_to_bf16_kernel<<<(DMODEL * DMODEL) / 2048, 256, 0, stream>>>(wv, wvbf);
    f32_to_bf16_kernel<<<(DMODEL * DMODEL) / 2048, 256, 0, stream>>>(wo, wobf);

    dim3 gemm_grid(DMODEL / 128, NROWS / 128);  // (16, 64)

    gemm_wmma_bf16<<<gemm_grid, 256, 0, stream>>>(
        xbf, wqbf, bq, qbuf, NROWS, DMODEL, DMODEL);
    gemm_wmma_bf16<<<gemm_grid, 256, 0, stream>>>(
        xbf, wkbf, bk, kbuf, NROWS, DMODEL, DMODEL);
    gemm_wmma_bf16<<<gemm_grid, 256, 0, stream>>>(
        xbf, wvbf, bv, vbuf, NROWS, DMODEL, DMODEL);

    rmsnorm_kernel<true><<<NROWS, 256, 0, stream>>>(qbuf, qnw, mix);
    rmsnorm_kernel<false><<<NROWS, 256, 0, stream>>>(kbuf, knw, nullptr);

    attn_kernel<<<(NROWS * NHEADS) / 8, 256, 0, stream>>>(qbuf, kbuf, vbuf,
                                                          attnbf);

    gemm_wmma_bf16<<<gemm_grid, 256, 0, stream>>>(
        attnbf, wobf, bo, out, NROWS, DMODEL, DMODEL);
}
